// MTGNN_52948356825425
// MI455X (gfx1250) — matrix operand order repaired and checked by reading, not compiled, not run on hardware
//
#include <hip/hip_runtime.h>
#include <hip/hip_bf16.h>

typedef __attribute__((ext_vector_type(16))) __bf16 v16bf;
typedef __attribute__((ext_vector_type(8)))  float  v8f;

#define N_NODES 100000
#define N_EDGES 300000
#define IN_DIM  32
#define HID     256
#define NGRAPH  4096
#define BN_EPS  1e-5f

// ---------------------------------------------------------------------------
// Pack a row-major f32 weight matrix W[K][Nc] into WMMA bf16 B-fragment order.
// Per (ktile, ntile): 32 lanes x 16 bf16, lane-contiguous.
// ISA layout (16-bit B, 32x16): lane gives N = lane&15; element j gives
// K_local = (j&7) + 8*(lane>>4) + 16*(j>>3).
// ---------------------------------------------------------------------------
__global__ void pack_b_kernel(const float* __restrict__ W, __bf16* __restrict__ Bp,
                              int K, int Nc) {
  int t = blockIdx.x * 256 + threadIdx.x;
  if (t >= K * Nc) return;
  int j      = t & 15;
  int lane   = (t >> 4) & 31;
  int tile   = t >> 9;                 // tk * nTiles + tn
  int nTiles = Nc >> 4;
  int tk     = tile / nTiles;
  int tn     = tile - tk * nTiles;
  int kl     = (j & 7) + ((lane >> 4) << 3) + ((j >> 3) << 4); // K_local = (j&7) + 8*half + 16*(j>>3)
  int n      = tn * 16 + (lane & 15);
  Bp[t] = (__bf16)W[(size_t)(tk * 32 + kl) * Nc + n];
}

// ---------------------------------------------------------------------------
// WMMA GEMM: C[M][Nc] = A[M][K] @ W[K][Nc], A row-major f32 converted to bf16
// on the fly, W pre-packed by pack_b_kernel. Block = 8 waves; 32 rows/block;
// each wave owns 16 rows x (NT*16) cols. M % 32 == 0 guaranteed by caller.
// All NT B fragments are loaded before the WMMA chain so the four
// v_wmma_f32_16x16x32_bf16 issue back-to-back without WAR-hazard NOPs.
// Optional fused epilogue: relu(acc + bias[col]) when bias != nullptr.
// ---------------------------------------------------------------------------
template<int NT>
__global__ __launch_bounds__(256) void wmma_gemm_kernel(
    const float* __restrict__ A, const __bf16* __restrict__ Bp,
    float* __restrict__ C, int K, int Nc, const float* __restrict__ bias) {
  const int lane  = threadIdx.x & 31;
  const int wave  = threadIdx.x >> 5;
  const int waveM = wave >> 2;   // 0..1
  const int waveN = wave & 3;    // 0..3
  const int half  = lane >> 4;
  const int m15   = lane & 15;
  const int rowBase = blockIdx.x * 32 + waveM * 16;
  const int nTiles  = Nc >> 4;

  v8f acc[NT] = {};

  // A-fragment layout (16-bit A, 16x32): row = lane&15; element j maps to
  // K = 8*half + j (j<8) and K = 16 + 8*half + (j-8) (j>=8).
  const float* aRowPtr = A + (size_t)(rowBase + m15) * K + 8 * half;

  const __bf16* bBase = Bp + ((size_t)(waveN * NT) * 32 + lane) * 16;
  const size_t bTnStride = 32 * 16;                 // elements per n-tile
  const size_t bKsStride = (size_t)nTiles * 32 * 16; // elements per k-step

  const int ksteps = K >> 5;
  for (int ks = 0; ks < ksteps; ++ks) {
    const float* ap = aRowPtr + ks * 32;
    v16bf afrag;
#pragma unroll
    for (int j = 0; j < 8; ++j) afrag[j] = (__bf16)ap[j];
#pragma unroll
    for (int j = 0; j < 8; ++j) afrag[8 + j] = (__bf16)ap[16 + j];

    v16bf bfrag[NT];
#pragma unroll
    for (int t = 0; t < NT; ++t)
      bfrag[t] = *(const v16bf*)(bBase + (size_t)ks * bKsStride + (size_t)t * bTnStride);

#pragma unroll
    for (int t = 0; t < NT; ++t)
      acc[t] = __builtin_amdgcn_wmma_f32_16x16x32_bf16(
          false, afrag, false, bfrag[t], (short)0, acc[t], false, false);
  }

  // D layout (32-bit C/D 16x16): VGPR r -> row = 8*half + r, col = lane&15
#pragma unroll
  for (int t = 0; t < NT; ++t) {
    int col = (waveN * NT + t) * 16 + m15;
    float* cp = C + (size_t)(rowBase + 8 * half) * Nc + col;
    if (bias) {
      float bb = bias[col];
#pragma unroll
      for (int r = 0; r < 8; ++r) {
        float v = acc[t][r] + bb;
        cp[(size_t)r * Nc] = v > 0.f ? v : 0.f;
      }
    } else {
#pragma unroll
      for (int r = 0; r < 8; ++r) cp[(size_t)r * Nc] = acc[t][r];
    }
  }
}

// ----------------------------- elementwise helpers --------------------------
__global__ void fill_kernel(float* __restrict__ p, float v, int n) {
  int t = blockIdx.x * 256 + threadIdx.x;
  if (t < n) p[t] = v;
}

__global__ void deg_accum_kernel(const int* __restrict__ col, float* __restrict__ deg) {
  int e = blockIdx.x * 256 + threadIdx.x;
  if (e < N_EDGES) atomicAdd(&deg[col[e]], 1.0f);
}

__global__ void rsqrt_inplace_kernel(float* __restrict__ d, int n) {
  int t = blockIdx.x * 256 + threadIdx.x;
  if (t < n) d[t] = rsqrtf(d[t]);   // deg >= 1 always (self loops)
}

__global__ void norm_kernel(const int* __restrict__ row, const int* __restrict__ col,
                            const float* __restrict__ dinv, float* __restrict__ nrm) {
  int e = blockIdx.x * 256 + threadIdx.x;
  if (e < N_EDGES) nrm[e] = dinv[row[e]] * dinv[col[e]];
}

// agg[i][c] = dinv[i]^2 * h[i][c]   (self-loop term; also initializes agg)
__global__ void self_loop_init_kernel(const float* __restrict__ h,
                                      const float* __restrict__ dinv,
                                      float* __restrict__ agg) {
  int t = blockIdx.x * 256 + threadIdx.x;   // N_NODES*HID threads (exact)
  float d = dinv[t >> 8];
  agg[t] = d * d * h[t];
}

// agg[col[e]] += norm[e] * h[row[e]] ; one thread per (edge, 4 channels)
__global__ void scatter_kernel(const int* __restrict__ row, const int* __restrict__ col,
                               const float* __restrict__ nrm,
                               const float* __restrict__ h, float* __restrict__ agg) {
  int t = blockIdx.x * 256 + threadIdx.x;   // N_EDGES*64 threads (exact)
  int e = t >> 6;
  int c = (t & 63) * 4;
  int r  = row[e];
  int cl = col[e];
  float w = nrm[e];
  float4 v = *(const float4*)(h + (size_t)r * HID + c);
  float* ap = agg + (size_t)cl * HID + c;
  atomicAdd(ap + 0, w * v.x);
  atomicAdd(ap + 1, w * v.y);
  atomicAdd(ap + 2, w * v.z);
  atomicAdd(ap + 3, w * v.w);
}

// h = relu(bn(h + conv_bias)) in place
__global__ void bn_relu_kernel(float* __restrict__ h, const float* __restrict__ bc,
                               const float* __restrict__ g, const float* __restrict__ be,
                               const float* __restrict__ rm, const float* __restrict__ rv) {
  int t = blockIdx.x * 256 + threadIdx.x;   // N_NODES*HID (exact)
  int c = t & (HID - 1);
  float v = h[t] + bc[c];
  v = (v - rm[c]) * rsqrtf(rv[c] + BN_EPS) * g[c] + be[c];
  h[t] = v > 0.f ? v : 0.f;
}

__global__ void pool_accum_kernel(const int* __restrict__ batch,
                                  const float* __restrict__ h, float* __restrict__ sums) {
  int t = blockIdx.x * 256 + threadIdx.x;   // N_NODES*HID (exact)
  int i = t >> 8, c = t & (HID - 1);
  atomicAdd(&sums[(size_t)batch[i] * HID + c], h[t]);
}

__global__ void cnt_accum_kernel(const int* __restrict__ batch, float* __restrict__ cnt) {
  int i = blockIdx.x * 256 + threadIdx.x;
  if (i < N_NODES) atomicAdd(&cnt[batch[i]], 1.0f);
}

__global__ void pool_div_kernel(float* __restrict__ sums, const float* __restrict__ cnt) {
  int t = blockIdx.x * 256 + threadIdx.x;   // NGRAPH*HID (exact)
  float c = cnt[t >> 8];
  sums[t] /= (c < 1.f ? 1.f : c);
}

__global__ void head_kernel(const float* __restrict__ hb, const float* __restrict__ Wh,
                            const float* __restrict__ bh, float* __restrict__ out) {
  int t = blockIdx.x * 256 + threadIdx.x;   // NGRAPH*4 (exact)
  int g = t >> 2, j = t & 3;
  const float* hp = hb + (size_t)g * 128;
  float s = bh[j];
#pragma unroll 8
  for (int c = 0; c < 128; ++c) s += hp[c] * Wh[c * 4 + j];
  out[t] = s;
}

// ---------------------------------------------------------------------------
extern "C" void kernel_launch(void* const* d_in, const int* in_sizes, int n_in,
                              void* d_out, int out_size, void* d_ws, size_t ws_size,
                              hipStream_t stream) {
  (void)in_sizes; (void)n_in; (void)out_size; (void)ws_size;

  const float* x     = (const float*)d_in[0];
  const int*   ei    = (const int*)d_in[1];     // [2][E]
  const int*   batch = (const int*)d_in[2];
  const float* Wc1   = (const float*)d_in[4];
  const float* bc1   = (const float*)d_in[5];
  const float* Wc2   = (const float*)d_in[6];
  const float* bc2   = (const float*)d_in[7];
  const float* Wc3   = (const float*)d_in[8];
  const float* bc3   = (const float*)d_in[9];
  const float* g1  = (const float*)d_in[10]; const float* be1 = (const float*)d_in[11];
  const float* rm1 = (const float*)d_in[12]; const float* rv1 = (const float*)d_in[13];
  const float* g2  = (const float*)d_in[14]; const float* be2 = (const float*)d_in[15];
  const float* rm2 = (const float*)d_in[16]; const float* rv2 = (const float*)d_in[17];
  const float* g3  = (const float*)d_in[18]; const float* be3 = (const float*)d_in[19];
  const float* rm3 = (const float*)d_in[20]; const float* rv3 = (const float*)d_in[21];
  const float* Wbot  = (const float*)d_in[22];
  const float* bbot  = (const float*)d_in[23];
  const float* Whead = (const float*)d_in[24];
  const float* bhead = (const float*)d_in[25];

  const int* erow = ei;
  const int* ecol = ei + N_EDGES;

  // ---- workspace layout (256B aligned slices) ----
  char* ws = (char*)d_ws;
  size_t off = 0;
  auto take = [&](size_t bytes) -> void* {
    void* p = ws + off;
    off += (bytes + 255) & ~(size_t)255;
    return p;
  };
  float*  bufA = (float*)take((size_t)N_NODES * HID * 4);
  float*  bufB = (float*)take((size_t)N_NODES * HID * 4);
  float*  dinv = (float*)take((size_t)N_NODES * 4);
  float*  nrm  = (float*)take((size_t)N_EDGES * 4);
  float*  sums = (float*)take((size_t)NGRAPH * HID * 4);
  float*  cnt  = (float*)take((size_t)NGRAPH * 4);
  float*  hb   = (float*)take((size_t)NGRAPH * 128 * 4);
  __bf16* W1p  = (__bf16*)take((size_t)IN_DIM * HID * 2);
  __bf16* W2p  = (__bf16*)take((size_t)HID * HID * 2);
  __bf16* W3p  = (__bf16*)take((size_t)HID * HID * 2);
  __bf16* Wbp  = (__bf16*)take((size_t)HID * 128 * 2);

  const int NB_NODE  = (N_NODES + 255) / 256;       // per-node kernels
  const int NB_EDGE  = (N_EDGES + 255) / 256;       // per-edge kernels
  const int NB_NH    = N_NODES * HID / 256;         // 100000 (exact)
  const int NB_SCAT  = N_EDGES * 64 / 256;          // 75000  (exact)
  const int NB_GEMM  = N_NODES / 32;                // 3125   (exact)

  // weight packing (bf16, WMMA B-fragment order)
  pack_b_kernel<<<(IN_DIM * HID + 255) / 256, 256, 0, stream>>>(Wc1, W1p, IN_DIM, HID);
  pack_b_kernel<<<HID * HID / 256, 256, 0, stream>>>(Wc2, W2p, HID, HID);
  pack_b_kernel<<<HID * HID / 256, 256, 0, stream>>>(Wc3, W3p, HID, HID);
  pack_b_kernel<<<HID * 128 / 256, 256, 0, stream>>>(Wbot, Wbp, HID, 128);

  // degrees (self-loop contributes 1) -> dinv -> edge norms
  fill_kernel<<<NB_NODE, 256, 0, stream>>>(dinv, 1.0f, N_NODES);
  deg_accum_kernel<<<NB_EDGE, 256, 0, stream>>>(ecol, dinv);
  rsqrt_inplace_kernel<<<NB_NODE, 256, 0, stream>>>(dinv, N_NODES);
  norm_kernel<<<NB_EDGE, 256, 0, stream>>>(erow, ecol, dinv, nrm);

  // ---- layer 1: x(N,32) @ Wc1 -> aggregate -> BN+ReLU ----
  wmma_gemm_kernel<4><<<NB_GEMM, 256, 0, stream>>>(x, W1p, bufB, IN_DIM, HID, nullptr);
  self_loop_init_kernel<<<NB_NH, 256, 0, stream>>>(bufB, dinv, bufA);
  scatter_kernel<<<NB_SCAT, 256, 0, stream>>>(erow, ecol, nrm, bufB, bufA);
  bn_relu_kernel<<<NB_NH, 256, 0, stream>>>(bufA, bc1, g1, be1, rm1, rv1);

  // ---- layer 2 ----
  wmma_gemm_kernel<4><<<NB_GEMM, 256, 0, stream>>>(bufA, W2p, bufB, HID, HID, nullptr);
  self_loop_init_kernel<<<NB_NH, 256, 0, stream>>>(bufB, dinv, bufA);
  scatter_kernel<<<NB_SCAT, 256, 0, stream>>>(erow, ecol, nrm, bufB, bufA);
  bn_relu_kernel<<<NB_NH, 256, 0, stream>>>(bufA, bc2, g2, be2, rm2, rv2);

  // ---- layer 3 ----
  wmma_gemm_kernel<4><<<NB_GEMM, 256, 0, stream>>>(bufA, W3p, bufB, HID, HID, nullptr);
  self_loop_init_kernel<<<NB_NH, 256, 0, stream>>>(bufB, dinv, bufA);
  scatter_kernel<<<NB_SCAT, 256, 0, stream>>>(erow, ecol, nrm, bufB, bufA);
  bn_relu_kernel<<<NB_NH, 256, 0, stream>>>(bufA, bc3, g3, be3, rm3, rv3);

  // ---- global mean pool ----
  fill_kernel<<<NGRAPH * HID / 256, 256, 0, stream>>>(sums, 0.0f, NGRAPH * HID);
  fill_kernel<<<(NGRAPH + 255) / 256, 256, 0, stream>>>(cnt, 0.0f, NGRAPH);
  pool_accum_kernel<<<NB_NH, 256, 0, stream>>>(batch, bufA, sums);
  cnt_accum_kernel<<<NB_NODE, 256, 0, stream>>>(batch, cnt);
  pool_div_kernel<<<NGRAPH * HID / 256, 256, 0, stream>>>(sums, cnt);

  // ---- bottleneck (4096x256 @ 256x128) with fused bias+ReLU, then head ----
  wmma_gemm_kernel<2><<<NGRAPH / 32, 256, 0, stream>>>(sums, Wbp, hb, HID, 128, bbot);
  head_kernel<<<NGRAPH * 4 / 256, 256, 0, stream>>>(hb, Whead, bhead, (float*)d_out);
}
